// MultiHeadAttention_54296976556094
// MI455X (gfx1250) — matrix operand order repaired and checked by reading, not compiled, not run on hardware
//
#include <hip/hip_runtime.h>

// Problem constants (from reference)
#define D_H    1024
#define HEAD   16
#define D_ATTN 64
#define BZ     2
#define LEN    2048
// scale = 1/sqrt(D_ATTN) / (LAYER_IDX+1) = 1/(8*12)
#define SCALE_QK (1.0f / 96.0f)

typedef __attribute__((ext_vector_type(16))) _Float16 v16h;
typedef __attribute__((ext_vector_type(8)))  _Float16 v8h;
typedef __attribute__((ext_vector_type(8)))  float    v8f;
typedef __attribute__((ext_vector_type(4)))  float    v4f;
typedef __attribute__((ext_vector_type(4)))  unsigned u32x4;
typedef __attribute__((ext_vector_type(8)))  unsigned u32x8;

#define WMMA_F16(a, b, c) \
  __builtin_amdgcn_wmma_f32_16x16x32_f16(false, (a), false, (b), (short)0, (c), false, false)

// ---- fragment loaders --------------------------------------------------
// A-matrix 16x32 f16 (ISA 7.12.2): lane L holds row (L&15); K = base..base+7
// (elements 0..7) and base+16..base+23 (elements 8..15), base = (L>>4)*8.
static __device__ __forceinline__ v16h load_a_f32(const float* p) {
  const v4f* q = (const v4f*)p;
  v4f x0 = q[0], x1 = q[1], x2 = q[4], x3 = q[5];   // K+0..7, K+16..23
  v16h a;
#pragma unroll
  for (int i = 0; i < 4; ++i) {
    a[i]      = (_Float16)x0[i];
    a[4 + i]  = (_Float16)x1[i];
    a[8 + i]  = (_Float16)x2[i];
    a[12 + i] = (_Float16)x3[i];
  }
  return a;
}

static __device__ __forceinline__ v16h load_a_f16(const _Float16* p) {
  const v8h* q = (const v8h*)p;
  v8h lo = q[0], hi = q[2];                         // K+0..7, K+16..23
  v16h a;
#pragma unroll
  for (int i = 0; i < 8; ++i) { a[i] = lo[i]; a[8 + i] = hi[i]; }
  return a;
}

// B-matrix 32x16 f16: lane L holds column (L&15); K = base..base+7
// (elements 0..7) and base+16..base+23 (elements 8..15), base = (L>>4)*8.
static __device__ __forceinline__ v16h load_b_f16(const _Float16* p) {
  const v8h* q = (const v8h*)p;
  v8h lo = q[0], hi = q[2];                         // K+0..7, K+16..23
  v16h b;
#pragma unroll
  for (int i = 0; i < 8; ++i) { b[i] = lo[i]; b[8 + i] = hi[i]; }
  return b;
}

// ---- Tensor Data Mover: 2D tile -> LDS (ISA ch.8 D#, 2-operand form) ----
// dims/strides in elements (f16, data_size code 1 = 2 bytes).
static __device__ __forceinline__ void tdm_load_2d_f16(
    unsigned lds_byte_off, const void* gaddr,
    unsigned tensor_d0, unsigned tensor_d1,
    unsigned tile_d0, unsigned tile_d1, unsigned stride0) {
  const unsigned long long ga = (unsigned long long)gaddr;
  u32x4 g0;
  g0[0] = 1u;                                        // count=1, user descriptor
  g0[1] = lds_byte_off;                              // lds_addr [63:32]
  g0[2] = (unsigned)ga;                              // global_addr [95:64]
  g0[3] = (unsigned)((ga >> 32) & 0x1FFFFFFull)      // global_addr [120:96]
        | 0x80000000u;                               // type=2 ("image") [127:126]
  u32x8 g1;
  g1[0] = 0x00010000u;                               // data_size=1 (2B), no mask
  g1[1] = (tensor_d0 & 0xFFFFu) << 16;               // tensor_dim0 [79:48] lo
  g1[2] = (tensor_d0 >> 16) | ((tensor_d1 & 0xFFFFu) << 16);
  g1[3] = (tensor_d1 >> 16) | (tile_d0 << 16);       // tile_dim0 [127:112]
  g1[4] = tile_d1;                                   // tile_dim1; tile_dim2=0
  g1[5] = stride0;                                   // tensor_dim0_stride lo
  g1[6] = 0u;                                        // stride0 hi | dim1_stride lo
  g1[7] = 0u;
  // VADDR2/VADDR3 omitted (NULL) -> <=2D tensor descriptor.
  asm volatile("tensor_load_to_lds %0, %1" :: "s"(g0), "s"(g1) : "memory");
}

static __device__ __forceinline__ unsigned lds_off(const void* p) {
  // generic pointers to LDS carry the LDS byte offset in the low 32 bits
  return (unsigned)(unsigned long long)p;
}

// ---- kernel 1: weight fp32 -> f16 transpose (Wt[n][k] = (f16)W[k][n]) ---
__global__ __launch_bounds__(256) void wcvt_kernel(const float* __restrict__ W,
                                                   _Float16* __restrict__ Wt) {
  __shared__ _Float16 t[32][33];
  const int tx = threadIdx.x & 31, ty = threadIdx.x >> 5;   // 32x8
  const int bx = blockIdx.x * 32;   // n tile
  const int by = blockIdx.y * 32;   // k tile
#pragma unroll
  for (int i = 0; i < 32; i += 8)
    t[ty + i][tx] = (_Float16)W[(size_t)(by + ty + i) * D_H + bx + tx];
  __syncthreads();
#pragma unroll
  for (int i = 0; i < 32; i += 8)
    Wt[(size_t)(bx + ty + i) * D_H + by + tx] = t[tx][ty + i];
}

// ---- kernel 2: projection GEMM, Y = X @ W (via Wt), f16 head-split out --
// 32x64 per-wave tile, software-pipelined fragment loads.
// mode 0/1: out[b][h][l][d]; mode 2: out[b][h][d][l] (V transposed)
__global__ __launch_bounds__(256) void proj_kernel(const float* __restrict__ X,
                                                   const _Float16* __restrict__ Wt,
                                                   _Float16* __restrict__ out,
                                                   int mode, float scale) {
  const int wave = threadIdx.x >> 5;
  const int lane = threadIdx.x & 31;
  const int nl = lane & 15, hi = lane >> 4;
  const int r0 = blockIdx.x * 256 + wave * 32;   // 8 waves x 32 rows
  const int c0 = blockIdx.y * 64;                // 4 col fragments per wave

  v8f acc0[4] = {v8f{}, v8f{}, v8f{}, v8f{}};
  v8f acc1[4] = {v8f{}, v8f{}, v8f{}, v8f{}};
  const float*    arow0 = X  + (size_t)(r0 + nl) * D_H + hi * 8;
  const float*    arow1 = arow0 + (size_t)16 * D_H;
  const _Float16* bcol  = Wt + (size_t)(c0 + nl) * D_H + hi * 8;

  v16h a0 = load_a_f32(arow0);
  v16h a1 = load_a_f32(arow1);
  v16h b[4];
#pragma unroll
  for (int d = 0; d < 4; ++d) b[d] = load_b_f16(bcol + (size_t)(d * 16) * D_H);

  for (int kk = 0; kk < D_H; kk += 32) {
    const int kn = (kk + 32 < D_H) ? kk + 32 : 0;   // clamped prefetch index
    v16h na0 = load_a_f32(arow0 + kn);
    v16h na1 = load_a_f32(arow1 + kn);
    v16h nb[4];
#pragma unroll
    for (int d = 0; d < 4; ++d) nb[d] = load_b_f16(bcol + (size_t)(d * 16) * D_H + kn);
#pragma unroll
    for (int d = 0; d < 4; ++d) acc0[d] = WMMA_F16(a0, b[d], acc0[d]);
#pragma unroll
    for (int d = 0; d < 4; ++d) acc1[d] = WMMA_F16(a1, b[d], acc1[d]);
    a0 = na0; a1 = na1;
#pragma unroll
    for (int d = 0; d < 4; ++d) b[d] = nb[d];
  }
#pragma unroll
  for (int g = 0; g < 2; ++g) {
#pragma unroll
    for (int d = 0; d < 4; ++d) {
#pragma unroll
      for (int r = 0; r < 8; ++r) {
        const float av = (g == 0) ? acc0[d][r] : acc1[d][r];
        const int gr = r0 + g * 16 + r + 8 * hi;   // C layout: M=r+8*(lane>=16)
        const int gc = c0 + d * 16 + nl;           //           N=lane&15
        const int bb = gr >> 11, l = gr & (LEN - 1);
        const int h  = gc >> 6,  dd = gc & 63;
        const _Float16 v = (_Float16)(av * scale);
        if (mode == 2)
          out[((size_t)(bb * HEAD + h) * D_ATTN + dd) * LEN + l] = v;
        else
          out[((size_t)(bb * HEAD + h) * LEN + l) * D_ATTN + dd] = v;
      }
    }
  }
}

// ---- kernel 3: flash attention; K/V tiles staged by TDM into LDS --------
__global__ __launch_bounds__(256) void attn_kernel(const _Float16* __restrict__ Qh,
                                                   const _Float16* __restrict__ Kh,
                                                   const _Float16* __restrict__ Vt,
                                                   _Float16* __restrict__ O16) {
  // per-wave: double-buffered K tile (32 keys x 64 d) and V tile (64 d x 32
  // keys), plus P shuffle buffer. Buffers are wave-private -> no barriers;
  // TDM completion is tracked with the wave's own TENSORcnt.
  __shared__ _Float16 kbuf[8][2][32 * 64];
  __shared__ _Float16 vbuf[8][2][64 * 32];
  __shared__ _Float16 pbuf[8][16 * 32];
  const int wave = threadIdx.x >> 5;
  const int lane = threadIdx.x & 31;
  const int nl = lane & 15, hi = lane >> 4;
  const int bh = blockIdx.y;
  const int b = bh >> 4, h = bh & 15;
  const int q0 = (blockIdx.x * 8 + wave) * 16;

  const _Float16* Qb = Qh + (size_t)bh * LEN * D_ATTN;
  const _Float16* Kb = Kh + (size_t)bh * LEN * D_ATTN;
  const _Float16* Vb = Vt + (size_t)bh * D_ATTN * LEN;
  _Float16* pw = pbuf[wave];

  // Q fragments (K dim = d_attn = 64 -> two 16x32 A frags), scale pre-folded
  const v16h aQ0 = load_a_f16(Qb + (size_t)(q0 + nl) * D_ATTN + hi * 8);
  const v16h aQ1 = load_a_f16(Qb + (size_t)(q0 + nl) * D_ATTN + 32 + hi * 8);

  v8f accO[4] = {v8f{}, v8f{}, v8f{}, v8f{}};
  float mrow[8], lrow[8];
#pragma unroll
  for (int r = 0; r < 8; ++r) { mrow[r] = -3.0e38f; lrow[r] = 0.0f; }

  const int nsteps = (q0 >> 5) + 1;              // covers keys 0..q0+15
  // prologue: DMA first K/V tiles
  tdm_load_2d_f16(lds_off(&kbuf[wave][0][0]), Kb, D_ATTN, LEN, D_ATTN, 32, D_ATTN);
  tdm_load_2d_f16(lds_off(&vbuf[wave][0][0]), Vb, LEN,    D_ATTN, 32, D_ATTN, LEN);

  for (int s = 0; s < nsteps; ++s) {
    const int kb = s * 32;
    const _Float16* kt = &kbuf[wave][s & 1][0];
    const _Float16* vt = &vbuf[wave][s & 1][0];
    if (s + 1 < nsteps) {
      const int kb2 = kb + 32;
      tdm_load_2d_f16(lds_off(&kbuf[wave][(s + 1) & 1][0]), Kb + (size_t)kb2 * D_ATTN,
                      D_ATTN, LEN - kb2, D_ATTN, 32, D_ATTN);
      tdm_load_2d_f16(lds_off(&vbuf[wave][(s + 1) & 1][0]), Vb + kb2,
                      LEN - kb2, D_ATTN, 32, D_ATTN, LEN);
      __builtin_amdgcn_s_wait_tensorcnt(2);      // current tiles landed
    } else {
      __builtin_amdgcn_s_wait_tensorcnt(0);
    }

    // S = Q K^T : two key-fragments x two d-halves, B frags from LDS K tile
    v8f sf[2] = {v8f{}, v8f{}};
#pragma unroll
    for (int f = 0; f < 2; ++f) {
      const _Float16* kp = kt + (f * 16 + nl) * D_ATTN + hi * 8;
      v16h bK0 = load_b_f16(kp);
      v16h bK1 = load_b_f16(kp + 32);
      sf[f] = WMMA_F16(aQ0, bK0, sf[f]);
      sf[f] = WMMA_F16(aQ1, bK1, sf[f]);
    }
    // online softmax (causal mask analytic); every row has >=1 valid column
    // per step since kb <= q0 <= row.
    float p0[8], p1[8];
#pragma unroll
    for (int r = 0; r < 8; ++r) {
      const int grow = q0 + r + 8 * hi;
      float s0 = (kb + nl      <= grow) ? sf[0][r] : -3.0e38f;
      float s1 = (kb + 16 + nl <= grow) ? sf[1][r] : -3.0e38f;
      float mx = fmaxf(s0, s1);
#pragma unroll
      for (int off = 8; off > 0; off >>= 1)
        mx = fmaxf(mx, __shfl_xor(mx, off, 16));
      const float mnew = fmaxf(mrow[r], mx);
      const float corr = __expf(mrow[r] - mnew);
      const float e0 = __expf(s0 - mnew);
      const float e1 = __expf(s1 - mnew);
      float rs = e0 + e1;
#pragma unroll
      for (int off = 8; off > 0; off >>= 1)
        rs += __shfl_xor(rs, off, 16);
      lrow[r] = lrow[r] * corr + rs;
      mrow[r] = mnew;
      p0[r] = e0; p1[r] = e1;
#pragma unroll
      for (int d = 0; d < 4; ++d) accO[d][r] *= corr;
    }
    // P: C-layout -> A-layout via per-wave LDS round trip (DS ops from the
    // same wave are in order; wait before the loads' data is consumed)
#pragma unroll
    for (int r = 0; r < 8; ++r) {
      pw[(r + 8 * hi) * 32 + nl]      = (_Float16)p0[r];
      pw[(r + 8 * hi) * 32 + 16 + nl] = (_Float16)p1[r];
    }
    __asm__ volatile("s_wait_dscnt 0" ::: "memory");
    const v16h aP = load_a_f16(pw + nl * 32 + hi * 8);
    // O += P @ V  (B frags from LDS V tile: [d][key], contiguous keys)
#pragma unroll
    for (int d = 0; d < 4; ++d) {
      v16h bV = load_b_f16(vt + (d * 16 + nl) * 32 + hi * 8);
      accO[d] = WMMA_F16(aP, bV, accO[d]);
    }
  }
  // normalize and write O in [b][l][h*64+d] f16 for the output projection
#pragma unroll
  for (int r = 0; r < 8; ++r) {
    const float inv = 1.0f / lrow[r];
    const int l = q0 + r + 8 * hi;
#pragma unroll
    for (int d = 0; d < 4; ++d)
      O16[((size_t)(b * LEN + l)) * D_H + h * D_ATTN + d * 16 + nl] =
          (_Float16)(accO[d][r] * inv);
  }
}

// ---- kernel 4: output projection, f32 out + bias (32x64 tile, pipelined) -
__global__ __launch_bounds__(256) void out_proj_kernel(const _Float16* __restrict__ O16,
                                                       const _Float16* __restrict__ WtO,
                                                       const float* __restrict__ bo,
                                                       float* __restrict__ out) {
  const int wave = threadIdx.x >> 5;
  const int lane = threadIdx.x & 31;
  const int nl = lane & 15, hi = lane >> 4;
  const int r0 = blockIdx.x * 256 + wave * 32;
  const int c0 = blockIdx.y * 64;

  v8f acc0[4] = {v8f{}, v8f{}, v8f{}, v8f{}};
  v8f acc1[4] = {v8f{}, v8f{}, v8f{}, v8f{}};
  const _Float16* arow0 = O16 + (size_t)(r0 + nl) * D_H + hi * 8;
  const _Float16* arow1 = arow0 + (size_t)16 * D_H;
  const _Float16* bcol  = WtO + (size_t)(c0 + nl) * D_H + hi * 8;

  v16h a0 = load_a_f16(arow0);
  v16h a1 = load_a_f16(arow1);
  v16h b[4];
#pragma unroll
  for (int d = 0; d < 4; ++d) b[d] = load_b_f16(bcol + (size_t)(d * 16) * D_H);

  for (int kk = 0; kk < D_H; kk += 32) {
    const int kn = (kk + 32 < D_H) ? kk + 32 : 0;
    v16h na0 = load_a_f16(arow0 + kn);
    v16h na1 = load_a_f16(arow1 + kn);
    v16h nb[4];
#pragma unroll
    for (int d = 0; d < 4; ++d) nb[d] = load_b_f16(bcol + (size_t)(d * 16) * D_H + kn);
#pragma unroll
    for (int d = 0; d < 4; ++d) acc0[d] = WMMA_F16(a0, b[d], acc0[d]);
#pragma unroll
    for (int d = 0; d < 4; ++d) acc1[d] = WMMA_F16(a1, b[d], acc1[d]);
    a0 = na0; a1 = na1;
#pragma unroll
    for (int d = 0; d < 4; ++d) b[d] = nb[d];
  }
#pragma unroll
  for (int g = 0; g < 2; ++g) {
#pragma unroll
    for (int d = 0; d < 4; ++d) {
#pragma unroll
      for (int r = 0; r < 8; ++r) {
        const float av = (g == 0) ? acc0[d][r] : acc1[d][r];
        const int gr = r0 + g * 16 + r + 8 * hi;
        const int gc = c0 + d * 16 + nl;
        out[(size_t)gr * D_H + gc] = av + bo[gc];
      }
    }
  }
}

// ---- launch -------------------------------------------------------------
extern "C" void kernel_launch(void* const* d_in, const int* in_sizes, int n_in,
                              void* d_out, int out_size, void* d_ws, size_t ws_size,
                              hipStream_t stream) {
  (void)in_sizes; (void)n_in; (void)out_size; (void)ws_size;
  const float* v  = (const float*)d_in[0];
  const float* k  = (const float*)d_in[1];
  const float* q  = (const float*)d_in[2];
  // d_in[3] is the causal mask; applied analytically in attn_kernel
  // (saves 33 MB of HBM reads).
  const float* Wv = (const float*)d_in[4];
  const float* Wk = (const float*)d_in[5];
  const float* Wq = (const float*)d_in[6];
  const float* Wo = (const float*)d_in[7];
  const float* bo = (const float*)d_in[8];
  float* out = (float*)d_out;

  // workspace: 4x 4M-f16 tensors + 4x 1M-f16 weights = 40 MB
  _Float16* ws = (_Float16*)d_ws;
  const size_t SZH = (size_t)BZ * HEAD * LEN * D_ATTN;   // 4M elements
  _Float16* Qh  = ws;
  _Float16* Kh  = ws + SZH;
  _Float16* Vt  = ws + 2 * SZH;
  _Float16* O16 = ws + 3 * SZH;
  _Float16* WtQ = ws + 4 * SZH;
  _Float16* WtK = WtQ + (size_t)D_H * D_H;
  _Float16* WtV = WtK + (size_t)D_H * D_H;
  _Float16* WtO = WtV + (size_t)D_H * D_H;

  dim3 tgrid(D_H / 32, D_H / 32);
  wcvt_kernel<<<tgrid, 256, 0, stream>>>(Wq, WtQ);
  wcvt_kernel<<<tgrid, 256, 0, stream>>>(Wk, WtK);
  wcvt_kernel<<<tgrid, 256, 0, stream>>>(Wv, WtV);
  wcvt_kernel<<<tgrid, 256, 0, stream>>>(Wo, WtO);

  dim3 pgrid(BZ * LEN / 256, D_H / 64);
  proj_kernel<<<pgrid, 256, 0, stream>>>(q, WtQ, Qh, 0, SCALE_QK);
  proj_kernel<<<pgrid, 256, 0, stream>>>(k, WtK, Kh, 1, 1.0f);
  proj_kernel<<<pgrid, 256, 0, stream>>>(v, WtV, Vt, 2, 1.0f);

  dim3 agrid(LEN / (16 * 8), BZ * HEAD);
  attn_kernel<<<agrid, 256, 0, stream>>>(Qh, Kh, Vt, O16);

  out_proj_kernel<<<pgrid, 256, 0, stream>>>(O16, WtO, bo, out);
}
